// CRFLayer_36498632081569
// MI455X (gfx1250) — compile-verified, matrix operand order checked by reading, not run on hardware
//
#include <hip/hip_runtime.h>
#include <hip/hip_bf16.h>

// ---------------------------------------------------------------------------
// BiLSTM(2 layer, bidir) + CRF for MI455X (gfx1250, wave32, WMMA).
// - Projection GEMMs: v_wmma_f32_16x16x32_bf16 with W tiles staged into LDS
//   via global_load_async_to_lds_b128 (ASYNCcnt / s_wait_asynccnt).
// - Recurrence: persistent 1-WGP kernel, h double-buffered in LDS,
//   column-tile-serialized to keep WMMA accumulators spill-free.
// - CRF: one wave32 per batch row (K=32 lanes), shuffle-based logsumexp.
// ---------------------------------------------------------------------------

#define Tn 512
#define Bn 128
#define En 256
#define Hn 256        // per-direction hidden
#define G4H 1024      // 4*H gate width
#define HIDn 512
#define Kn 32
#define START_TAG 30
#define END_TAG 31
#define NEGV (-100000.0f)

typedef __attribute__((ext_vector_type(16))) __bf16       v16bf;
typedef __attribute__((ext_vector_type(8)))  float        v8f;
typedef __attribute__((ext_vector_type(4)))  unsigned int v4u;

union FragU { v16bf v; v4u q[2]; };

__device__ __forceinline__ float sigf(float x)  { return 1.f / (1.f + __expf(-x)); }
__device__ __forceinline__ float tanhfast(float x){ return 2.f * sigf(2.f * x) - 1.f; }

// Load a 16x32 bf16 WMMA fragment (A layout; B uses same pattern with W stored
// row-major [N][K], i.e. column n of B is row n of W).
// lane<16 -> K {k0+0..7, k0+16..23}; lane>=16 -> K {k0+8..15, k0+24..31}
__device__ __forceinline__ v16bf load_frag(const __bf16* __restrict__ base,
                                           int row0, int ldk, int k0, int lane) {
  int r  = row0 + (lane & 15);
  int ks = (lane >> 4) << 3;
  const __bf16* p = base + (size_t)r * ldk + k0 + ks;
  FragU f;
  f.q[0] = *(const v4u*)(p);
  f.q[1] = *(const v4u*)(p + 16);
  return f.v;
}

// ---------------------------------------------------------------------------
// fp32 -> bf16 conversion (weights, one-time)
// ---------------------------------------------------------------------------
__global__ void f32_to_bf16(const float* __restrict__ s, __bf16* __restrict__ d, int n) {
  int i = blockIdx.x * blockDim.x + threadIdx.x;
  if (i < n) d[i] = (__bf16)s[i];
}

// ---------------------------------------------------------------------------
// Embedding gather: x[t][b][e] = bf16(emb[words[b][t]][e])
// ---------------------------------------------------------------------------
__global__ void gather_embed(const int* __restrict__ words, const float* __restrict__ emb,
                             __bf16* __restrict__ x, int total) {
  int i = blockIdx.x * blockDim.x + threadIdx.x;
  if (i >= total) return;
  int e  = i & (En - 1);
  int tb = i >> 8;            // /E
  int b  = tb & (Bn - 1);
  int t  = tb >> 7;           // /B
  int w  = words[b * Tn + t];
  x[i] = (__bf16)emb[(size_t)w * En + e];
}

// ---------------------------------------------------------------------------
// Projection GEMM: C[M][N] = A[M][Kd] * W[N][Kd]^T, fp32 out.
// Block = 8 waves covering 128 rows x 64 cols; the shared 64x32 W chunk is
// staged into LDS with async-to-LDS loads (double buffered, one chunk ahead).
// ---------------------------------------------------------------------------
__global__ __launch_bounds__(256) void wmma_gemm_stage(
    const __bf16* __restrict__ A, const __bf16* __restrict__ W,
    float* __restrict__ C, int M, int N, int Kd)
{
  __shared__ __bf16 wt[2][64 * 32];     // 2 x 4KB double buffer

  int tid  = threadIdx.x;
  int wave = tid >> 5, lane = tid & 31;
  int colGroups = N >> 6;
  int bc = blockIdx.x % colGroups;
  int br = blockIdx.x / colGroups;
  int row0 = (br * 8 + wave) << 4;
  int n0   = bc << 6;
  if (row0 >= M) return;

  // staging: thread i moves 16B of W row (n0+i/4), k-chunk (i%4)*8
  int sr = tid >> 2;                    // 0..63
  int sk = (tid & 3) << 3;              // 0,8,16,24
  unsigned lds0 = (unsigned)(size_t)&wt[0][sr * 32 + sk];  // low 32b == LDS offset
  unsigned lds1 = (unsigned)(size_t)&wt[1][sr * 32 + sk];
  const __bf16* wsrc = W + (size_t)(n0 + sr) * Kd + sk;

  const int nk = Kd >> 5;
  asm volatile("global_load_async_to_lds_b128 %0, %1, off"
               :: "v"(lds0), "v"((unsigned long long)(size_t)wsrc) : "memory");

  const v8f vzero = {};
  v8f acc[4];
#pragma unroll
  for (int i = 0; i < 4; ++i) acc[i] = vzero;

  for (int i = 0; i < nk; ++i) {
    asm volatile("s_wait_asynccnt 0x0" ::: "memory");  // own chunk i landed
    __syncthreads();                                   // everyone's chunk i landed
    if (i + 1 < nk) {                                  // prefetch chunk i+1
      unsigned dst = (i & 1) ? lds0 : lds1;
      asm volatile("global_load_async_to_lds_b128 %0, %1, off"
                   :: "v"(dst),
                      "v"((unsigned long long)(size_t)(wsrc + ((i + 1) << 5)))
                   : "memory");
    }
    const __bf16* wtile = wt[i & 1];
    v16bf a = load_frag(A, row0, Kd, i << 5, lane);
#pragma unroll
    for (int ct = 0; ct < 4; ++ct) {
      v16bf b = load_frag(wtile, ct << 4, 32, 0, lane);
      acc[ct] = __builtin_amdgcn_wmma_f32_16x16x32_bf16(
          false, a, false, b, (short)0, acc[ct], false, false);
    }
  }

  int lm = lane & 15, lh = lane >> 4;
#pragma unroll
  for (int ct = 0; ct < 4; ++ct) {
    int n = n0 + (ct << 4) + lm;
#pragma unroll
    for (int v = 0; v < 8; ++v) {
      int m = row0 + v + (lh << 3);
      C[(size_t)m * N + n] = acc[ct][v];
    }
  }
}

// ---------------------------------------------------------------------------
// Small GEMM for the tag head (N=32): per-wave global loads, + bias.
// ---------------------------------------------------------------------------
template<int CT>
__global__ __launch_bounds__(256) void wmma_gemm_bias(
    const __bf16* __restrict__ A, const __bf16* __restrict__ W,
    const float* __restrict__ bias, float* __restrict__ C,
    int M, int N, int Kd)
{
  int gtid = blockIdx.x * blockDim.x + threadIdx.x;
  int wave = gtid >> 5;
  int lane = gtid & 31;
  int colGroups = N / (16 * CT);
  int rowTile = wave / colGroups;
  int cg      = wave - rowTile * colGroups;
  int row0 = rowTile << 4;
  if (row0 >= M) return;

  const v8f vzero = {};
  v8f acc[CT];
#pragma unroll
  for (int i = 0; i < CT; ++i) acc[i] = vzero;

  for (int k0 = 0; k0 < Kd; k0 += 32) {
    v16bf a = load_frag(A, row0, Kd, k0, lane);
#pragma unroll
    for (int ct = 0; ct < CT; ++ct) {
      v16bf b = load_frag(W, (cg * CT + ct) << 4, Kd, k0, lane);
      acc[ct] = __builtin_amdgcn_wmma_f32_16x16x32_bf16(
          false, a, false, b, (short)0, acc[ct], false, false);
    }
  }

  int lm = lane & 15, lh = lane >> 4;
#pragma unroll
  for (int ct = 0; ct < CT; ++ct) {
    int n = ((cg * CT + ct) << 4) + lm;
    float bv = bias ? bias[n] : 0.f;
#pragma unroll
    for (int v = 0; v < 8; ++v) {
      int m = row0 + v + (lh << 3);
      C[(size_t)m * N + n] = acc[ct][v] + bv;
    }
  }
}

// ---------------------------------------------------------------------------
// LSTM recurrence (one direction): persistent single workgroup (32 waves).
// gates[t] = xproj[t] + h[t-1] @ Whh^T + bias ; i,f,g,o cell.
// Column tiles processed serially so only acc[4] (one tile/gate) is live ->
// no VGPR spills. h double-buffered in LDS; h streamed to hout (bf16).
// ---------------------------------------------------------------------------
__global__ __launch_bounds__(1024) void lstm_recur(
    const float*  __restrict__ xproj,  // [T][B][1024] fp32
    const __bf16* __restrict__ Whh,    // [1024][256] bf16 (L2-resident)
    const float*  __restrict__ bias,   // [1024]
    __bf16*       __restrict__ hout,   // [T][B][512]; write at column dirOff
    int dirOff, int reverse)
{
  extern __shared__ char smem_raw[];
  __bf16* hb0 = (__bf16*)smem_raw;     // [128][256]
  __bf16* hb1 = hb0 + Bn * Hn;

  int tid  = threadIdx.x;
  int wave = tid >> 5, lane = tid & 31;
  int lm = lane & 15, lh = lane >> 4;
  int rowTile = wave >> 2;             // 0..7  (batch tile)
  int ctBase  = (wave & 3) << 2;       // 0,4,8,12 (H col-tile base)

  for (int i = tid; i < Bn * Hn; i += 1024) hb0[i] = (__bf16)0.f;
  __syncthreads();

  const v8f vzero = {};
  v8f c[4];
#pragma unroll
  for (int i = 0; i < 4; ++i) c[i] = vzero;

  for (int step = 0; step < Tn; ++step) {
    int t = reverse ? (Tn - 1 - step) : step;
    const __bf16* hcur = (step & 1) ? hb1 : hb0;
    __bf16*       hnxt = (step & 1) ? hb0 : hb1;

    const float* xp = xproj + (size_t)t * Bn * G4H;
    __bf16* ho = hout + (size_t)t * Bn * HIDn + dirOff;
    if (step + 1 < Tn) {  // prefetch next timestep's projections
      int tnext = reverse ? (Tn - 2 - step) : (step + 1);
      __builtin_prefetch(xproj + (size_t)tnext * Bn * G4H + tid * 16, 0, 1);
    }

#pragma unroll
    for (int ct = 0; ct < 4; ++ct) {
      int colT = ctBase + ct;
      v8f acc[4];
#pragma unroll
      for (int g = 0; g < 4; ++g) acc[g] = vzero;

#pragma unroll 1
      for (int k0 = 0; k0 < Hn; k0 += 32) {
        v16bf a = load_frag(hcur, rowTile << 4, Hn, k0, lane);
#pragma unroll
        for (int g = 0; g < 4; ++g) {
          v16bf bf = load_frag(Whh, g * Hn + (colT << 4), Hn, k0, lane);
          acc[g] = __builtin_amdgcn_wmma_f32_16x16x32_bf16(
              false, a, false, bf, (short)0, acc[g], false, false);
        }
      }

      int nH = (colT << 4) + lm;                      // 0..255
      float bi = bias[nH], bff = bias[Hn + nH];
      float bg = bias[2 * Hn + nH], bo = bias[3 * Hn + nH];
#pragma unroll
      for (int v = 0; v < 8; ++v) {
        int m = (rowTile << 4) + v + (lh << 3);       // batch row 0..127
        const float* xr = xp + (size_t)m * G4H + nH;
        float gi = acc[0][v] + xr[0]      + bi;
        float gf = acc[1][v] + xr[Hn]     + bff;
        float gg = acc[2][v] + xr[2 * Hn] + bg;
        float go = acc[3][v] + xr[3 * Hn] + bo;
        float cn = sigf(gf) * c[ct][v] + sigf(gi) * tanhfast(gg);
        c[ct][v] = cn;
        float hv = sigf(go) * tanhfast(cn);
        hnxt[m * Hn + nH]         = (__bf16)hv;
        ho[(size_t)m * HIDn + nH] = (__bf16)hv;
      }
    }
    __syncthreads();   // step s writes visible before step s+1 reads
  }
}

// ---------------------------------------------------------------------------
// CRF: gold path score (one thread per batch row)
// ---------------------------------------------------------------------------
__global__ void crf_gold(const int* __restrict__ labels, const float* __restrict__ emit,
                         const float* __restrict__ trans, float* __restrict__ gold) {
  int b = blockIdx.x * blockDim.x + threadIdx.x;
  if (b >= Bn) return;
  float g = 0.f; int prev = START_TAG;
  for (int t = 0; t < Tn; ++t) {
    int cur = labels[b * Tn + t];
    g += trans[cur * Kn + prev] + emit[((size_t)t * Bn + b) * Kn + cur];
    prev = cur;
  }
  g += trans[END_TAG * Kn + prev];
  gold[b] = g;
}

// ---------------------------------------------------------------------------
// CRF log-partition: one wave per batch row, lane k = tag k (K==32==wave32).
// ---------------------------------------------------------------------------
__global__ __launch_bounds__(128) void crf_partition(
    const float* __restrict__ emit, const float* __restrict__ trans,
    float* __restrict__ logZ)
{
  int gtid = blockIdx.x * blockDim.x + threadIdx.x;
  int b = gtid >> 5, k = gtid & 31;
  if (b >= Bn) return;
  float tr[Kn];
#pragma unroll
  for (int j = 0; j < Kn; ++j) tr[j] = trans[k * Kn + j];
  float dp = (k == START_TAG) ? 0.f : NEGV;
  for (int t = 0; t < Tn; ++t) {
    float e = emit[((size_t)t * Bn + b) * Kn + k];
    float m = -3.0e38f;
#pragma unroll
    for (int j = 0; j < Kn; ++j) m = fmaxf(m, tr[j] + __shfl(dp, j, 32));
    float s = 0.f;
#pragma unroll
    for (int j = 0; j < Kn; ++j) s += __expf(tr[j] + __shfl(dp, j, 32) - m);
    dp = e + m + __logf(s);
  }
  float v = dp + trans[END_TAG * Kn + k];
  float m = v;
#pragma unroll
  for (int off = 16; off; off >>= 1) m = fmaxf(m, __shfl_xor(m, off, 32));
  float s = __expf(v - m);
#pragma unroll
  for (int off = 16; off; off >>= 1) s += __shfl_xor(s, off, 32);
  if (k == 0) logZ[b] = m + __logf(s);
}

// ---------------------------------------------------------------------------
// CRF Viterbi: same wave layout; u8 backpointers; ties -> lowest index.
// ---------------------------------------------------------------------------
__global__ __launch_bounds__(128) void crf_viterbi(
    const float* __restrict__ emit, const float* __restrict__ trans,
    unsigned char* __restrict__ bps, int* __restrict__ best)
{
  int gtid = blockIdx.x * blockDim.x + threadIdx.x;
  int b = gtid >> 5, k = gtid & 31;
  if (b >= Bn) return;
  float tr[Kn];
#pragma unroll
  for (int j = 0; j < Kn; ++j) tr[j] = trans[k * Kn + j];
  float dp = (k == START_TAG) ? 0.f : NEGV;
  for (int t = 0; t < Tn; ++t) {
    float e = emit[((size_t)t * Bn + b) * Kn + k];
    float m = -3.0e38f; int am = 0;
#pragma unroll
    for (int j = 0; j < Kn; ++j) {
      float val = tr[j] + __shfl(dp, j, 32);
      if (val > m) { m = val; am = j; }
    }
    bps[((size_t)t * Bn + b) * Kn + k] = (unsigned char)am;
    dp = e + m;
  }
  float v = dp + trans[END_TAG * Kn + k];
  int bk = k;
#pragma unroll
  for (int off = 16; off; off >>= 1) {
    float ov = __shfl_xor(v, off, 32);
    int   ok = __shfl_xor(bk, off, 32);
    if (ov > v || (ov == v && ok < bk)) { v = ov; bk = ok; }
  }
  if (k == 0) best[b] = bk;
}

__global__ void nll_reduce(const float* __restrict__ logZ, const float* __restrict__ gold,
                           float* __restrict__ out) {
  __shared__ float sh[Bn];
  int i = threadIdx.x;
  sh[i] = logZ[i] - gold[i];
  __syncthreads();
  for (int s = Bn / 2; s; s >>= 1) { if (i < s) sh[i] += sh[i + s]; __syncthreads(); }
  if (i == 0) out[0] = sh[0] / (float)Bn;
}

__global__ void backtrace(const unsigned char* __restrict__ bps, const int* __restrict__ best,
                          float* __restrict__ out) {
  int b = blockIdx.x * blockDim.x + threadIdx.x;
  if (b >= Bn) return;
  int tag = best[b];
  for (int t = Tn - 1; t >= 0; --t) {
    out[1 + b * Tn + t] = (float)tag;
    tag = (int)bps[((size_t)t * Bn + b) * Kn + tag];
  }
}

// ---------------------------------------------------------------------------
// Host-side orchestration
// ---------------------------------------------------------------------------
static inline char* ws_alloc(char*& p, size_t bytes) {
  char* r = p;
  p += (bytes + 255) & ~(size_t)255;
  return r;
}

extern "C" void kernel_launch(void* const* d_in, const int* in_sizes, int n_in,
                              void* d_out, int out_size, void* d_ws, size_t ws_size,
                              hipStream_t stream) {
  const int*   words  = (const int*)  d_in[0];
  const int*   labels = (const int*)  d_in[1];
  const float* emb    = (const float*)d_in[2];
  const float* Wih0f  = (const float*)d_in[3];
  const float* Whh0f  = (const float*)d_in[4];
  const float* b0f    = (const float*)d_in[5];
  const float* Wih0b  = (const float*)d_in[6];
  const float* Whh0b  = (const float*)d_in[7];
  const float* b0b    = (const float*)d_in[8];
  const float* Wih1f  = (const float*)d_in[9];
  const float* Whh1f  = (const float*)d_in[10];
  const float* b1f    = (const float*)d_in[11];
  const float* Wih1b  = (const float*)d_in[12];
  const float* Whh1b  = (const float*)d_in[13];
  const float* b1b    = (const float*)d_in[14];
  const float* Wtag   = (const float*)d_in[15];
  const float* btag   = (const float*)d_in[16];
  const float* trans  = (const float*)d_in[17];
  float* out = (float*)d_out;

  const size_t M = (size_t)Tn * Bn;          // 65536 rows
  char* p = (char*)d_ws;
  __bf16* x     = (__bf16*)ws_alloc(p, M * En * 2);
  float*  xproj = (float*) ws_alloc(p, M * G4H * 4);          // reused per dir-layer
  __bf16* h1    = (__bf16*)ws_alloc(p, M * HIDn * 2);
  __bf16* h2    = (__bf16*)ws_alloc(p, M * HIDn * 2);
  float*  emitb = (float*) ws_alloc(p, M * Kn * 4);
  __bf16* wih0f = (__bf16*)ws_alloc(p, (size_t)G4H * En * 2);
  __bf16* whh0f = (__bf16*)ws_alloc(p, (size_t)G4H * Hn * 2);
  __bf16* wih0b = (__bf16*)ws_alloc(p, (size_t)G4H * En * 2);
  __bf16* whh0b = (__bf16*)ws_alloc(p, (size_t)G4H * Hn * 2);
  __bf16* wih1f = (__bf16*)ws_alloc(p, (size_t)G4H * HIDn * 2);
  __bf16* whh1f = (__bf16*)ws_alloc(p, (size_t)G4H * Hn * 2);
  __bf16* wih1b = (__bf16*)ws_alloc(p, (size_t)G4H * HIDn * 2);
  __bf16* whh1b = (__bf16*)ws_alloc(p, (size_t)G4H * Hn * 2);
  __bf16* wtagb = (__bf16*)ws_alloc(p, (size_t)Kn * HIDn * 2);
  unsigned char* bps = (unsigned char*)ws_alloc(p, M * Kn);
  float* logZ = (float*)ws_alloc(p, Bn * 4);
  float* gold = (float*)ws_alloc(p, Bn * 4);
  int*   best = (int*)  ws_alloc(p, Bn * 4);

  // --- weight conversion (fp32 -> bf16) ---
  struct { const float* s; __bf16* d; int n; } cv[] = {
    {Wih0f, wih0f, G4H * En}, {Whh0f, whh0f, G4H * Hn},
    {Wih0b, wih0b, G4H * En}, {Whh0b, whh0b, G4H * Hn},
    {Wih1f, wih1f, G4H * HIDn}, {Whh1f, whh1f, G4H * Hn},
    {Wih1b, wih1b, G4H * HIDn}, {Whh1b, whh1b, G4H * Hn},
    {Wtag,  wtagb, Kn * HIDn},
  };
  for (auto& c : cv)
    f32_to_bf16<<<(c.n + 255) / 256, 256, 0, stream>>>(c.s, c.d, c.n);

  // --- embedding gather ---
  {
    int total = (int)(M * En);
    gather_embed<<<(total + 255) / 256, 256, 0, stream>>>(words, emb, x, total);
  }

  const size_t lds_bytes = 2u * Bn * Hn * sizeof(__bf16);   // 128 KB double buffer
  const int projBlocks = (int)((M / 128) * (G4H / 64));     // 8192

  // --- layer 0 forward / backward ---
  wmma_gemm_stage<<<projBlocks, 256, 0, stream>>>(x, wih0f, xproj, (int)M, G4H, En);
  lstm_recur<<<1, 1024, lds_bytes, stream>>>(xproj, whh0f, b0f, h1, 0, 0);
  wmma_gemm_stage<<<projBlocks, 256, 0, stream>>>(x, wih0b, xproj, (int)M, G4H, En);
  lstm_recur<<<1, 1024, lds_bytes, stream>>>(xproj, whh0b, b0b, h1, Hn, 1);
  // --- layer 1 forward / backward ---
  wmma_gemm_stage<<<projBlocks, 256, 0, stream>>>(h1, wih1f, xproj, (int)M, G4H, HIDn);
  lstm_recur<<<1, 1024, lds_bytes, stream>>>(xproj, whh1f, b1f, h2, 0, 0);
  wmma_gemm_stage<<<projBlocks, 256, 0, stream>>>(h1, wih1b, xproj, (int)M, G4H, HIDn);
  lstm_recur<<<1, 1024, lds_bytes, stream>>>(xproj, whh1b, b1b, h2, Hn, 1);

  // --- tag projection (N=32) ---
  {
    long waves = (long)(M / 16);   // colGroups == 1 for CT=2
    int blocks = (int)((waves * 32 + 255) / 256);
    wmma_gemm_bias<2><<<blocks, 256, 0, stream>>>(h2, wtagb, btag, emitb,
                                                  (int)M, Kn, HIDn);
  }

  // --- CRF ---
  crf_gold<<<1, 128, 0, stream>>>(labels, emitb, trans, gold);
  crf_partition<<<(Bn * 32 + 127) / 128, 128, 0, stream>>>(emitb, trans, logZ);
  crf_viterbi<<<(Bn * 32 + 127) / 128, 128, 0, stream>>>(emitb, trans, bps, best);
  nll_reduce<<<1, Bn, 0, stream>>>(logZ, gold, out);
  backtrace<<<1, 128, 0, stream>>>(bps, best, out);

  (void)in_sizes; (void)n_in; (void)out_size; (void)ws_size;
}
